// MultiHeadAttention_63539746177188
// MI455X (gfx1250) — compile-verified, matrix operand order checked by reading, not run on hardware
//
#include <hip/hip_runtime.h>
#include <hip/hip_bf16.h>

typedef __bf16 bf16_t;
typedef __attribute__((ext_vector_type(16))) __bf16 v16bf;
typedef __attribute__((ext_vector_type(8)))  __bf16 v8bf;
typedef __attribute__((ext_vector_type(4)))  __bf16 v4bf;
typedef __attribute__((ext_vector_type(8)))  float  v8f;
typedef __attribute__((ext_vector_type(4)))  float  v4f;

namespace {
constexpr int B  = 4;
constexpr int S  = 2048;
constexpr int D  = 1024;
constexpr int H  = 16;
constexpr int DK = 64;
constexpr int M  = B * S;      // 8192 rows in every GEMM
constexpr int NK = D;          // N = K = 1024 in every GEMM
constexpr int BM = 64, BN = 64, BK = 32;
constexpr int PITCH = 72;      // bf16 elems per LDS row: 144B (16B-aligned, conflict-free)
constexpr float SCALE = 0.125f; // 1/sqrt(DK)
}

__device__ __forceinline__ v16bf make_frag(const bf16_t* lo, const bf16_t* hi) {
    union { v16bf v; v8bf h[2]; } u;
    u.h[0] = *(const v8bf*)lo;   // K = kg .. kg+7      -> elements 0..7
    u.h[1] = *(const v8bf*)hi;   // K = kg+16 .. kg+23  -> elements 8..15
    return u.v;
}

__device__ __forceinline__ v8f wmma_bf16(v16bf a, v16bf b, v8f c) {
    return __builtin_amdgcn_wmma_f32_16x16x32_bf16(
        /*neg_a=*/false, a, /*neg_b=*/false, b,
        /*c_mod=*/(short)0, c, /*reuse_a=*/false, /*reuse_b=*/false);
}

// Async global -> LDS 16B copy (ASYNCcnt-tracked, CDNA5 §15.18 op 98).
// lds_off = LDS byte address (low 32 bits of flat shared pointer).
__device__ __forceinline__ void async_copy_b128(unsigned lds_off, const void* gaddr) {
    asm volatile("global_load_async_to_lds_b128 %0, %1, off"
                 :: "v"(lds_off), "v"(gaddr)
                 : "memory");
}
__device__ __forceinline__ void wait_async0() {
    asm volatile("s_wait_asynccnt 0x0" ::: "memory");
}

// C[m,n] = sum_k A[m,k] * W[n,k] + bias[n]     (A: [M,NK] fp32, W: [NK,NK] fp32)
// EPI 0: fp32 store to [M,NK]
// EPI 1: bf16 store of val*SCALE to [M,NK]          (q, scale folded)
// EPI 2: bf16 store to [M,NK]                        (k)
// EPI 3: bf16 store transposed to [B, NK, S]         (v^T, K-major for P@V)
template <int EPI>
__global__ __launch_bounds__(128)
void gemm_kernel(const float* __restrict__ A, const float* __restrict__ W,
                 const float* __restrict__ bias, void* __restrict__ outp)
{
    __shared__ bf16_t lA[BM * PITCH];
    __shared__ bf16_t lB[BN * PITCH];

    const int m0   = blockIdx.x * BM;
    const int n0   = blockIdx.y * BN;
    const int tid  = threadIdx.x;
    const int lane = tid & 31;
    const int wave = tid >> 5;          // 4 waves, each owns 16 rows x 64 cols
    const int fr   = lane & 15;
    const int hg   = lane >> 4;         // half-wave group
    const int kg   = hg << 3;           // 0 or 8: K-chunk base inside fragment

    v8f acc[4] = { {}, {}, {}, {} };

    for (int kk = 0; kk < NK; kk += BK) {
        // Stage 64x32 fp32 tiles of A and W into LDS as bf16 (coalesced b128 loads)
        #pragma unroll
        for (int i = 0; i < 4; ++i) {
            const int lin = i * 128 + tid;        // 0..511
            const int row = lin >> 3;             // 8 float4 per 32-wide row
            const int c4  = (lin & 7) << 2;
            v4f a4 = *(const v4f*)(A + (size_t)(m0 + row) * NK + kk + c4);
            v4f w4 = *(const v4f*)(W + (size_t)(n0 + row) * NK + kk + c4);
            if (kk + BK < NK)   // hint next K tile into cache (global_prefetch_b8)
                __builtin_prefetch(A + (size_t)(m0 + row) * NK + kk + BK + c4, 0, 0);
            v4bf ab, wb;
            #pragma unroll
            for (int e = 0; e < 4; ++e) { ab[e] = (bf16_t)a4[e]; wb[e] = (bf16_t)w4[e]; }
            *(v4bf*)&lA[row * PITCH + c4] = ab;
            *(v4bf*)&lB[row * PITCH + c4] = wb;
        }
        __syncthreads();

        const v16bf af = make_frag(&lA[(wave * 16 + fr) * PITCH + kg],
                                   &lA[(wave * 16 + fr) * PITCH + kg + 16]);
        #pragma unroll
        for (int jn = 0; jn < 4; ++jn) {
            const v16bf bfg = make_frag(&lB[(jn * 16 + fr) * PITCH + kg],
                                        &lB[(jn * 16 + fr) * PITCH + kg + 16]);
            acc[jn] = wmma_bf16(af, bfg, acc[jn]);
        }
        __syncthreads();
    }

    // Epilogue: D-layout row = r + 8*hg, col = fr (per 16x16 tile)
    #pragma unroll
    for (int jn = 0; jn < 4; ++jn) {
        const int n = n0 + jn * 16 + fr;
        const float bv = bias[n];
        #pragma unroll
        for (int r = 0; r < 8; ++r) {
            const int mrow = m0 + wave * 16 + r + (hg << 3);
            const float v = acc[jn][r] + bv;
            if (EPI == 0) {
                ((float*)outp)[(size_t)mrow * NK + n] = v;
            } else if (EPI == 1) {
                ((bf16_t*)outp)[(size_t)mrow * NK + n] = (bf16_t)(v * SCALE);
            } else if (EPI == 2) {
                ((bf16_t*)outp)[(size_t)mrow * NK + n] = (bf16_t)v;
            } else {
                const int bb = mrow / S, ss = mrow % S;
                ((bf16_t*)outp)[((size_t)bb * NK + n) * S + ss] = (bf16_t)v;
            }
        }
    }
}

// One WG per (b, h, 64 q-rows); 4 waves, each owns 16 q-rows.
// Scores are computed TRANSPOSED (S^T = K @ Q^T): the WMMA D-layout then holds
// q-row per lane / j along regs, which IS the A-fragment layout for P @ V.
// Pass 1: online (max,sum) per lane + one shfl_xor(16) merge. No global writes.
// Pass 2: recompute S^T via WMMA, finalize p, write attn fp32 once (contiguous
//         32B per lane), feed p directly into ctx WMMA. K / V^T chunks staged in
//         LDS with async global->LDS copies shared by all 4 waves.
__global__ __launch_bounds__(128)
void attn_kernel(const bf16_t* __restrict__ qb, const bf16_t* __restrict__ kb,
                 const bf16_t* __restrict__ vt, float* __restrict__ attn,
                 float* __restrict__ ctx)
{
    __shared__ bf16_t lK[32 * PITCH];   // K chunk: 32 j-rows x 64 d
    __shared__ bf16_t lV[64 * PITCH];   // V^T chunk: 64 d-rows x 32 j

    const int qt   = blockIdx.x;        // 0..31 (q tile of 64 rows)
    const int h    = blockIdx.y;
    const int b    = blockIdx.z;
    const int tid  = threadIdx.x;
    const int lane = tid & 31;
    const int wave = tid >> 5;
    const int fr   = lane & 15;
    const int hg   = lane >> 4;
    const int kg   = hg << 3;

    // Q fragments (B-operand of S^T): lane holds q-row m=fr of this wave's strip
    const bf16_t* qrow = qb + ((size_t)b * S + qt * 64 + wave * 16 + fr) * D + h * DK;
    const v16bf qf0 = make_frag(qrow + kg,      qrow + kg + 16);
    const v16bf qf1 = make_frag(qrow + 32 + kg, qrow + 32 + kg + 16);

    const bf16_t* kbase = kb + ((size_t)b * S) * D + h * DK;   // + j*D + d
    const bf16_t* vbase = vt + ((size_t)b * D + h * DK) * S;   // + d*S + j

    float rmax = -3.0e38f, rsum = 0.0f;

    // ---------------- Pass 1: softmax stats only ----------------
    for (int j0 = 0; j0 < S; j0 += 32) {
        #pragma unroll
        for (int i = 0; i < 2; ++i) {                 // 32 rows x 128B = 256 x 16B
            const int c = i * 128 + tid;
            const int row = c >> 3, c16 = (c & 7) << 3;
            async_copy_b128((unsigned)(size_t)&lK[row * PITCH + c16],
                            kbase + (size_t)(j0 + row) * D + c16);
        }
        wait_async0();
        __syncthreads();

        #pragma unroll
        for (int u = 0; u < 2; ++u) {
            const bf16_t* kr = &lK[(u * 16 + fr) * PITCH];
            v8f st = {};
            st = wmma_bf16(make_frag(kr + kg,      kr + kg + 16),      qf0, st);
            st = wmma_bf16(make_frag(kr + 32 + kg, kr + 32 + kg + 16), qf1, st);
            #pragma unroll
            for (int r = 0; r < 8; ++r) {
                const float sv = st[r];
                const float mn = fmaxf(rmax, sv);
                rsum = rsum * __expf(rmax - mn) + __expf(sv - mn);
                rmax = mn;
            }
        }
        __syncthreads();
    }

    // Lanes L and L^16 hold the two j-halves of the same q-row: single merge.
    {
        const float om = __shfl_xor(rmax, 16, 32);
        const float os = __shfl_xor(rsum, 16, 32);
        const float mn = fmaxf(rmax, om);
        rsum = rsum * __expf(rmax - mn) + os * __expf(om - mn);
        rmax = mn;
    }
    const float inv = 1.0f / rsum;

    float* prow = attn + (((size_t)b * H + h) * S + qt * 64 + wave * 16 + fr) * S;

    v8f cacc[4] = { {}, {}, {}, {} };

    // ---------------- Pass 2: recompute, finalize p, ctx = P @ V ----------------
    for (int j0 = 0; j0 < S; j0 += 32) {
        #pragma unroll
        for (int i = 0; i < 2; ++i) {                 // K chunk
            const int c = i * 128 + tid;
            const int row = c >> 3, c16 = (c & 7) << 3;
            async_copy_b128((unsigned)(size_t)&lK[row * PITCH + c16],
                            kbase + (size_t)(j0 + row) * D + c16);
        }
        #pragma unroll
        for (int i = 0; i < 2; ++i) {                 // V^T chunk: 64 rows x 64B
            const int c = i * 128 + tid;
            const int row = c >> 2, c16 = (c & 3) << 3;
            async_copy_b128((unsigned)(size_t)&lV[row * PITCH + c16],
                            vbase + (size_t)row * S + j0 + c16);
        }
        wait_async0();
        __syncthreads();

        union { v16bf v; bf16_t e[16]; } pf;
        #pragma unroll
        for (int u = 0; u < 2; ++u) {
            const bf16_t* kr = &lK[(u * 16 + fr) * PITCH];
            v8f st = {};
            st = wmma_bf16(make_frag(kr + kg,      kr + kg + 16),      qf0, st);
            st = wmma_bf16(make_frag(kr + 32 + kg, kr + 32 + kg + 16), qf1, st);
            v4f p0, p1;
            #pragma unroll
            for (int r = 0; r < 8; ++r) {
                const float p = __expf(st[r] - rmax) * inv;
                if (r < 4) p0[r] = p; else p1[r - 4] = p;
                pf.e[u * 8 + r] = (bf16_t)p;          // A-frag element r (+8 for u=1)
            }
            // attn row m=fr, cols j0 + u*16 + kg + 0..7 : contiguous 32B store
            *(v4f*)(prow + j0 + u * 16 + kg)     = p0;
            *(v4f*)(prow + j0 + u * 16 + kg + 4) = p1;
        }

        #pragma unroll
        for (int dn = 0; dn < 4; ++dn) {
            const bf16_t* vr = &lV[(dn * 16 + fr) * PITCH];
            cacc[dn] = wmma_bf16(pf.v, make_frag(vr + kg, vr + kg + 16), cacc[dn]);
        }
        __syncthreads();
    }

    // ctx fp32 [B, S, D]; D-layout: col d=fr, row = r + 8*hg
    #pragma unroll
    for (int dn = 0; dn < 4; ++dn) {
        const int d = h * DK + dn * 16 + fr;
        #pragma unroll
        for (int r = 0; r < 8; ++r) {
            const int srow = qt * 64 + wave * 16 + r + hg * 8;
            ctx[((size_t)b * S + srow) * D + d] = cacc[dn][r];
        }
    }
}

extern "C" void kernel_launch(void* const* d_in, const int* in_sizes, int n_in,
                              void* d_out, int out_size, void* d_ws, size_t ws_size,
                              hipStream_t stream)
{
    (void)in_sizes; (void)n_in; (void)out_size; (void)ws_size;
    const float* Q  = (const float*)d_in[0];
    const float* Ki = (const float*)d_in[1];
    const float* V  = (const float*)d_in[2];
    const float* wq = (const float*)d_in[3];
    const float* bq = (const float*)d_in[4];
    const float* wk = (const float*)d_in[5];
    const float* bk = (const float*)d_in[6];
    const float* wv = (const float*)d_in[7];
    const float* bv = (const float*)d_in[8];
    const float* wo = (const float*)d_in[9];
    const float* bo = (const float*)d_in[10];

    float* out  = (float*)d_out;                 // [B,S,D]
    float* attn = out + (size_t)B * S * D;       // [B,H,S,S]

    // Workspace: q(bf16) | k(bf16) | v^T(bf16) | ctx(fp32)  -> 16+16+16+32 MB
    bf16_t* qbuf = (bf16_t*)d_ws;
    bf16_t* kbuf = qbuf + (size_t)M * D;
    bf16_t* vtb  = kbuf + (size_t)M * D;
    float*  ctx  = (float*)(vtb + (size_t)M * D);

    dim3 ggrid(M / BM, NK / BN);   // (128, 16)
    dim3 blk(128);

    gemm_kernel<1><<<ggrid, blk, 0, stream>>>(Q,  wq, bq, (void*)qbuf);
    gemm_kernel<2><<<ggrid, blk, 0, stream>>>(Ki, wk, bk, (void*)kbuf);
    gemm_kernel<3><<<ggrid, blk, 0, stream>>>(V,  wv, bv, (void*)vtb);

    dim3 agrid(S / 64, H, B);      // (32, 16, 4)
    attn_kernel<<<agrid, blk, 0, stream>>>(qbuf, kbuf, vtb, attn, ctx);

    gemm_kernel<0><<<ggrid, blk, 0, stream>>>(ctx, wo, bo, d_out);
}
